// SpatialProcessor_84035330113920
// MI455X (gfx1250) — compile-verified, matrix operand order checked by reading, not run on hardware
//
#include <hip/hip_runtime.h>
#include <cstdint>

typedef __attribute__((ext_vector_type(16))) _Float16 v16h;
typedef __attribute__((ext_vector_type(8)))  float    v8f;

// Problem constants (reference file).
constexpr int Bn  = 8;
constexpr int Nn  = 10000;
constexpr int Fin = 128;
constexpr int Un  = 32;
constexpr int H1  = 4;
constexpr int Mrows = Bn * Nn;   // 80000 GEMM rows (divisible by 16)

__device__ __forceinline__ float lrelu(float x) { return x > 0.f ? x : 0.2f * x; }

// Float atomic max via integer ordering trick (correct for mixed signs, -inf init).
__device__ __forceinline__ void atomicMaxF(float* addr, float val) {
  if (val >= 0.f) atomicMax((int*)addr, __float_as_int(val));
  else            atomicMin((unsigned int*)addr, __float_as_uint(val));
}

// ---------------------------------------------------------------------------
// WMMA GEMM:  D[M x NC] = f16(A') x f16(W),  A' = RELU_BIAS ? relu(A + bias[k]) : A
// A is [M x 128] f32 row-major, W is [128 x NC] f32 row-major.
// Block = 256 threads = 8 waves; each wave owns one 16-row tile, all NC cols.
// K = 128 -> 4 k-steps of 32 with v_wmma_f32_16x16x32_f16.
//
// W is staged into LDS PRE-SWIZZLED into B-fragment order:
//   wlds[ ((ks*CT + ct)*32 + lane)*16 + j ]  ==  B-frag element j of `lane`
// so each lane fetches its whole fragment with one 32-byte contiguous read
// (2x ds_load_b128) instead of 16x ds_load_u16.
// ---------------------------------------------------------------------------
template <int NC, bool RELU_BIAS>
__global__ __launch_bounds__(256)
void gemm_wmma(const float* __restrict__ A, const float* __restrict__ W,
               const float* __restrict__ bias, float* __restrict__ D, int M) {
  constexpr int CT = NC / 16;
  __shared__ __align__(32) _Float16 wlds[128 * NC];
  __shared__ float blds[128];
  const int tid = threadIdx.x;

  // Cooperative staging: gather W (f32) -> fragment-ordered f16 LDS image.
  for (int i = tid; i < 128 * NC; i += 256) {
    const int j    = i & 15;           // element within fragment
    const int l    = (i >> 4) & 31;    // lane
    const int tile = i >> 9;           // (ks, ct)
    const int ct   = tile % CT;
    const int ks   = tile / CT;
    const int kbl  = (l >> 4) * 8;     // per-lane K sub-offset
    const int K    = ks * 32 + kbl + (j < 8 ? j : 8 + j);
    const int c    = ct * 16 + (l & 15);
    wlds[i] = (_Float16)W[K * NC + c];
  }
  if (RELU_BIAS) { if (tid < 128) blds[tid] = bias[tid]; }
  __syncthreads();

  const int wave = tid >> 5;
  const int lane = tid & 31;
  const int mt   = blockIdx.x * 8 + wave;          // 16-row tile index
  if (mt * 16 >= M) return;

  const int mrow = mt * 16 + (lane & 15);          // A-fragment row for this lane
  const int kb   = (lane >> 4) * 8;                // K sub-offset per A layout

  v8f acc[CT];
#pragma unroll
  for (int c = 0; c < CT; ++c)
#pragma unroll
    for (int r = 0; r < 8; ++r) acc[c][r] = 0.0f;

#pragma unroll
  for (int ks = 0; ks < 4; ++ks) {
    const int kbase = ks * 32 + kb;
    // A fragment: two 8-float runs at kbase and kbase+16, as float4 vector loads.
    const float4* ap4 = (const float4*)(A + (size_t)mrow * 128 + kbase);
    float4 lo0 = ap4[0], lo1 = ap4[1];             // floats kbase+0..7
    float4 hi0 = ap4[4], hi1 = ap4[5];             // floats kbase+16..23
    float lov[8] = {lo0.x, lo0.y, lo0.z, lo0.w, lo1.x, lo1.y, lo1.z, lo1.w};
    float hiv[8] = {hi0.x, hi0.y, hi0.z, hi0.w, hi1.x, hi1.y, hi1.z, hi1.w};
    v16h afrag;
#pragma unroll
    for (int j = 0; j < 8; ++j) {
      float v0 = lov[j], v1 = hiv[j];
      if (RELU_BIAS) {
        v0 = fmaxf(v0 + blds[kbase + j], 0.0f);
        v1 = fmaxf(v1 + blds[kbase + 16 + j], 0.0f);
      }
      afrag[j]     = (_Float16)v0;
      afrag[8 + j] = (_Float16)v1;
    }
#pragma unroll
    for (int ct = 0; ct < CT; ++ct) {
      // Whole fragment in one aligned 32-byte LDS read.
      const v16h bfrag = *(const v16h*)&wlds[((ks * CT + ct) * 32 + lane) * 16];
      acc[ct] = __builtin_amdgcn_wmma_f32_16x16x32_f16(
          false, afrag, false, bfrag, (short)0, acc[ct], false, false);
    }
  }

  // C/D layout: lanes 0-15 hold M=r, lanes 16-31 hold M=8+r, N = lane&15.
  const int rbase = mt * 16 + (lane >> 4) * 8;
  const int col   = lane & 15;
#pragma unroll
  for (int ct = 0; ct < CT; ++ct) {
#pragma unroll
    for (int r = 0; r < 8; ++r)
      D[(size_t)(rbase + r) * NC + ct * 16 + col] = acc[ct][r];
  }
}

// ---------------------------------------------------------------------------
// Per-node attention logits: e_src[n][b][h] = dot(h[b,n,h,:], a_src[h,:])
// ---------------------------------------------------------------------------
template <int H>
__global__ void attn_logits(const float* __restrict__ ht,
                            const float* __restrict__ a_src,
                            const float* __restrict__ a_dst,
                            float* __restrict__ es, float* __restrict__ ed) {
  int q = blockIdx.x * blockDim.x + threadIdx.x;
  if (q >= Bn * Nn * H) return;
  const int b = q / (Nn * H);
  const int t = q % (Nn * H);
  const int n = t / H;
  const int h = t % H;
  const float* row = ht + ((size_t)(b * Nn + n) * H + h) * 32;
  float ss = 0.f, ds = 0.f;
#pragma unroll
  for (int u = 0; u < 32; ++u) {
    const float v = row[u];
    ss += v * a_src[h * 32 + u];
    ds += v * a_dst[h * 32 + u];
  }
  es[(size_t)n * (Bn * H) + b * H + h] = ss;
  ed[(size_t)n * (Bn * H) + b * H + h] = ds;
}

__global__ void fill_f32(float* __restrict__ p, float v, int n) {
  int i = blockIdx.x * blockDim.x + threadIdx.x;
  if (i < n) p[i] = v;
}

__global__ void init_out(float* __restrict__ out, const float* __restrict__ b2, int n) {
  int i = blockIdx.x * blockDim.x + threadIdx.x;
  if (i < n) out[i] = b2[i & 31];
}

// ---------------------------------------------------------------------------
// Edge pass A: segment max of leaky-relu logits at dst.
// ---------------------------------------------------------------------------
template <int H>
__global__ void edge_max(const int* __restrict__ src, const int* __restrict__ dst,
                         const float* __restrict__ es, const float* __restrict__ ed,
                         float* __restrict__ segmax, int nE) {
  const int HB = Bn * H;
  int q = blockIdx.x * blockDim.x + threadIdx.x;
  if (q >= nE * HB) return;
  const int e = q / HB, r = q % HB;
  const int s = src[e], d = dst[e];
  const float l = lrelu(es[(size_t)s * HB + r] + ed[(size_t)d * HB + r]);
  atomicMaxF(&segmax[(size_t)d * HB + r], l);
}

// ---------------------------------------------------------------------------
// Edge pass B: denom[dst] += exp(logit - segmax[dst])
// ---------------------------------------------------------------------------
template <int H>
__global__ void edge_denom(const int* __restrict__ src, const int* __restrict__ dst,
                           const float* __restrict__ es, const float* __restrict__ ed,
                           const float* __restrict__ segmax, float* __restrict__ denom,
                           int nE) {
  const int HB = Bn * H;
  int q = blockIdx.x * blockDim.x + threadIdx.x;
  if (q >= nE * HB) return;
  const int e = q / HB, r = q % HB;
  const int s = src[e], d = dst[e];
  const float l = lrelu(es[(size_t)s * HB + r] + ed[(size_t)d * HB + r]);
  float m = segmax[(size_t)d * HB + r];
  if (!__builtin_isfinite(m)) m = 0.0f;
  atomicAdd(&denom[(size_t)d * HB + r], __expf(l - m));
}

// ---------------------------------------------------------------------------
// Edge pass C: agg[dst] += alpha * h[src], float4-vectorized, fully coalesced
// (layer 1: one 256-thread block == one edge; layer 2: 4 edges per block).
// ---------------------------------------------------------------------------
template <int H>
__global__ void edge_aggregate(const int* __restrict__ src, const int* __restrict__ dst,
                               const float* __restrict__ es, const float* __restrict__ ed,
                               const float* __restrict__ segmax, const float* __restrict__ denom,
                               const float* __restrict__ ht, float* __restrict__ agg, int nE) {
  const int HB  = Bn * H;
  const int CPE = HB * 8;                        // float4 chunks per edge
  int q = blockIdx.x * blockDim.x + threadIdx.x;
  if (q >= nE * CPE) return;
  const int e  = q / CPE;
  const int r  = q % CPE;
  const int bh = r >> 3;                         // (b,h) pair
  const int u4 = r & 7;                          // float4 chunk within 32 units
  const int s = src[e], d = dst[e];

  const float l = lrelu(es[(size_t)s * HB + bh] + ed[(size_t)d * HB + bh]);
  float m = segmax[(size_t)d * HB + bh];
  if (!__builtin_isfinite(m)) m = 0.0f;
  const float alpha = __expf(l - m) / (denom[(size_t)d * HB + bh] + 1e-9f);

  const int b  = bh / H;
  const int hh = bh % H;
  const float4 v = *(const float4*)(ht + ((size_t)(b * Nn + s) * H + hh) * 32 + u4 * 4);
  float* drow = agg + ((size_t)(b * Nn + d) * H + hh) * 32 + u4 * 4;
  atomicAdd(drow + 0, alpha * v.x);
  atomicAdd(drow + 1, alpha * v.y);
  atomicAdd(drow + 2, alpha * v.z);
  atomicAdd(drow + 3, alpha * v.w);
}

// ---------------------------------------------------------------------------
extern "C" void kernel_launch(void* const* d_in, const int* in_sizes, int n_in,
                              void* d_out, int out_size, void* d_ws, size_t ws_size,
                              hipStream_t stream) {
  const float* x    = (const float*)d_in[0];
  const int*   eSrc = (const int*)  d_in[1];
  const int*   eDst = (const int*)  d_in[2];
  const float* W1   = (const float*)d_in[3];
  const float* a1s  = (const float*)d_in[4];
  const float* a1d  = (const float*)d_in[5];
  const float* b1   = (const float*)d_in[6];
  const float* W2   = (const float*)d_in[7];
  const float* a2s  = (const float*)d_in[8];
  const float* a2d  = (const float*)d_in[9];
  const float* b2   = (const float*)d_in[10];
  float* out = (float*)d_out;
  const int NE = in_sizes[1];                    // edges incl. self loops

  // Workspace carve-out (~99 MB of f32).
  float* ws  = (float*)d_ws;
  float* h1t  = ws; ws += (size_t)Mrows * 128;   // layer-1 transformed features
  float* agg1 = ws; ws += (size_t)Mrows * 128;   // layer-1 aggregation (pre relu+bias)
  float* h2t  = ws; ws += (size_t)Mrows * 32;    // layer-2 transformed features
  float* e1s  = ws; ws += (size_t)Nn * 32;
  float* e1d  = ws; ws += (size_t)Nn * 32;
  float* m1   = ws; ws += (size_t)Nn * 32;
  float* d1   = ws; ws += (size_t)Nn * 32;
  float* e2s  = ws; ws += (size_t)Nn * 8;
  float* e2d  = ws; ws += (size_t)Nn * 8;
  float* m2   = ws; ws += (size_t)Nn * 8;
  float* d2   = ws; ws += (size_t)Nn * 8;

  const int TB = 256;

  // ================= Layer 1 =================
  gemm_wmma<128, false><<<Mrows / (16 * 8), TB, 0, stream>>>(x, W1, nullptr, h1t, Mrows);
  attn_logits<H1><<<(Bn * Nn * H1 + TB - 1) / TB, TB, 0, stream>>>(h1t, a1s, a1d, e1s, e1d);
  fill_f32<<<(Nn * 32 + TB - 1) / TB, TB, 0, stream>>>(m1, -__builtin_inff(), Nn * 32);
  hipMemsetAsync(d1, 0, (size_t)Nn * 32 * sizeof(float), stream);
  hipMemsetAsync(agg1, 0, (size_t)Mrows * 128 * sizeof(float), stream);

  const int tA1 = NE * Bn * H1;
  edge_max<H1>  <<<(tA1 + TB - 1) / TB, TB, 0, stream>>>(eSrc, eDst, e1s, e1d, m1, NE);
  edge_denom<H1><<<(tA1 + TB - 1) / TB, TB, 0, stream>>>(eSrc, eDst, e1s, e1d, m1, d1, NE);
  const int tC1 = NE * (Bn * H1 * 8);            // 256 float4 chunks per edge
  edge_aggregate<H1><<<(tC1 + TB - 1) / TB, TB, 0, stream>>>(eSrc, eDst, e1s, e1d, m1, d1,
                                                             h1t, agg1, NE);

  // ================= Layer 2 =================
  // relu(agg1 + b1) fused into A-fragment conversion.
  gemm_wmma<32, true><<<Mrows / (16 * 8), TB, 0, stream>>>(agg1, W2, b1, h2t, Mrows);
  attn_logits<1><<<(Bn * Nn + TB - 1) / TB, TB, 0, stream>>>(h2t, a2s, a2d, e2s, e2d);
  fill_f32<<<(Nn * 8 + TB - 1) / TB, TB, 0, stream>>>(m2, -__builtin_inff(), Nn * 8);
  hipMemsetAsync(d2, 0, (size_t)Nn * 8 * sizeof(float), stream);
  init_out<<<(out_size + TB - 1) / TB, TB, 0, stream>>>(out, b2, out_size);

  const int tA2 = NE * Bn;
  edge_max<1>  <<<(tA2 + TB - 1) / TB, TB, 0, stream>>>(eSrc, eDst, e2s, e2d, m2, NE);
  edge_denom<1><<<(tA2 + TB - 1) / TB, TB, 0, stream>>>(eSrc, eDst, e2s, e2d, m2, d2, NE);
  const int tC2 = NE * (Bn * 8);                 // 64 float4 chunks per edge
  edge_aggregate<1><<<(tC2 + TB - 1) / TB, TB, 0, stream>>>(eSrc, eDst, e2s, e2d, m2, d2,
                                                            h2t, out, NE);
}
